// MultiHeadAttention_16836271800517
// MI455X (gfx1250) — matrix operand order repaired and checked by reading, not compiled, not run on hardware
//
#include <hip/hip_runtime.h>

// ---------------------------------------------------------------------------
// MI455X (gfx1250) fused multi-head attention:
//   1) convert x / w_qkv / w_proj f32 -> bf16 (once)
//   2) QKV GEMM via v_wmma_f32_16x16x32_bf16, bias fused, 1/sqrt(D) folded
//      into Q, scatter to [B,H,N,D] bf16 Q/K/V
//   2b) transpose V once: [B,H,N,D] -> [B,H,D,N] (LDS tile transpose)
//   3) flash attention: K/V^T tiles double-buffered in LDS via
//      global_load_async_to_lds_b128 + s_wait_asynccnt; online softmax;
//      S/P never touch HBM
//   4) projection GEMM -> f32 output
// All intermediates (~38 MB) are L2-resident (192 MB L2).
// ---------------------------------------------------------------------------

typedef __attribute__((ext_vector_type(16))) __bf16 v16bf;
typedef __attribute__((ext_vector_type(8)))  float  v8f;

union Vec16 { uint4 q[2]; v16bf v; };

constexpr int   kDim   = 384;
constexpr int   kF     = 1152;   // 3*kDim
constexpr int   kHeads = 6;
constexpr int   kHS    = 64;
constexpr int   kSeq   = 4096;
constexpr int   kBatch = 2;
constexpr int   kM     = kBatch * kSeq;  // 8192 rows
constexpr float kLog2e = 1.4426950408889634f;

__device__ __forceinline__ unsigned short f2bf(float x) {
  unsigned int u = __float_as_uint(x);
  u += 0x7fffu + ((u >> 16) & 1u);          // round-to-nearest-even
  return (unsigned short)(u >> 16);
}

__global__ void cvt_bf16_kernel(const float* __restrict__ s,
                                unsigned short* __restrict__ d, int n) {
  int i = blockIdx.x * blockDim.x + threadIdx.x;
  if (i < n) d[i] = f2bf(s[i]);
}

__device__ __forceinline__ v8f wmma_bf16(v16bf a, v16bf b, v8f c) {
  return __builtin_amdgcn_wmma_f32_16x16x32_bf16(false, a, false, b,
                                                 (short)0, c, false, false);
}

// async memory -> LDS copy, 16 bytes per lane, tracked with ASYNCcnt
__device__ __forceinline__ void async_b128(unsigned int lds_off,
                                           unsigned long long gaddr) {
  asm volatile("global_load_async_to_lds_b128 %0, %1, off"
               :: "v"(lds_off), "v"(gaddr) : "memory");
}

// ---------------------------------------------------------------------------
// Shared GEMM tile core: per-wave 32x32 output tile, K = 384, step 32.
// A layout (16-bit, 16x32): lane<16 -> e0..7 = K k..k+7,  e8..15 = K k+16..k+23
//                           lane>=16 -> e0..7 = K k+8..15, e8..15 = K k+24..31
// B layout (16-bit, 32x16): lanes 0-15 hold K k..k+7,k+16..23, lanes 16-31 K k+8..15,k+24..31 (as A),
//                           column = lane&15 -> contiguous row of W[f][*].
// ---------------------------------------------------------------------------
__device__ __forceinline__ void gemm_tile_384(const unsigned short* __restrict__ A,
                                              const unsigned short* __restrict__ W,
                                              int m0, int f0, int hh, int cc,
                                              v8f acc[2][2]) {
  const unsigned short* ar0 = A + (size_t)(m0 + cc) * kDim;
  const unsigned short* ar1 = ar0 + (size_t)16 * kDim;
  const unsigned short* wr0 = W + (size_t)(f0 + cc) * kDim;
  const unsigned short* wr1 = wr0 + (size_t)16 * kDim;
#pragma unroll 4
  for (int k = 0; k < kDim; k += 32) {
    Vec16 a0, a1, b0, b1;
    a0.q[0] = *(const uint4*)(ar0 + k + 8 * hh);
    a0.q[1] = *(const uint4*)(ar0 + k + 16 + 8 * hh);
    a1.q[0] = *(const uint4*)(ar1 + k + 8 * hh);
    a1.q[1] = *(const uint4*)(ar1 + k + 16 + 8 * hh);
    b0.q[0] = *(const uint4*)(wr0 + k + 8 * hh);
    b0.q[1] = *(const uint4*)(wr0 + k + 16 + 8 * hh);
    b1.q[0] = *(const uint4*)(wr1 + k + 8 * hh);
    b1.q[1] = *(const uint4*)(wr1 + k + 16 + 8 * hh);
    acc[0][0] = wmma_bf16(a0.v, b0.v, acc[0][0]);
    acc[0][1] = wmma_bf16(a0.v, b1.v, acc[0][1]);
    acc[1][0] = wmma_bf16(a1.v, b0.v, acc[1][0]);
    acc[1][1] = wmma_bf16(a1.v, b1.v, acc[1][1]);
  }
}

// ---------------------------------------------------------------------------
// Kernel 2: QKV projection. Block = 256 threads (8 waves) -> 128x64 tile.
// Writes Q (scaled by 1/8), K, V as bf16 in [B,H,N,D].
// ---------------------------------------------------------------------------
__global__ __launch_bounds__(256) void qkv_gemm_kernel(
    const unsigned short* __restrict__ xb, const unsigned short* __restrict__ wqb,
    const float* __restrict__ bias,
    unsigned short* __restrict__ qo, unsigned short* __restrict__ ko,
    unsigned short* __restrict__ vo) {
  const int lane = threadIdx.x & 31;
  const int wave = threadIdx.x >> 5;
  const int hh = lane >> 4, cc = lane & 15;
  const int m0 = blockIdx.x * 128 + (wave & 3) * 32;
  const int f0 = blockIdx.y * 64 + (wave >> 2) * 32;
  v8f acc[2][2] = {};
  gemm_tile_384(xb, wqb, m0, f0, hh, cc, acc);
#pragma unroll
  for (int ni = 0; ni < 2; ++ni) {
    const int f = f0 + ni * 16 + cc;
    const float bv = bias[f];
    const int which = f / kDim;
    const int c = f % kDim;
    const int h = c / kHS;
    const int d = c % kHS;
#pragma unroll
    for (int mi = 0; mi < 2; ++mi) {
#pragma unroll
      for (int r = 0; r < 8; ++r) {
        const int m = m0 + mi * 16 + r + 8 * hh;   // C/D layout: row = r + 8*half
        const int b = m >> 12;
        const int n = m & (kSeq - 1);
        const float val = acc[mi][ni][r] + bv;
        const size_t idx = ((size_t)((b * kHeads + h) * kSeq + n)) * kHS + d;
        if (which == 0)      qo[idx] = f2bf(val * 0.125f);  // 1/sqrt(64)
        else if (which == 1) ko[idx] = f2bf(val);
        else                 vo[idx] = f2bf(val);
      }
    }
  }
}

// ---------------------------------------------------------------------------
// Kernel 2b: one-time V transpose [B,H,N,D] -> [B,H,D,N] via LDS 64x64 tile.
// ---------------------------------------------------------------------------
__global__ __launch_bounds__(256) void vtrans_kernel(
    const unsigned short* __restrict__ v, unsigned short* __restrict__ vt) {
  __shared__ __align__(16) unsigned short t[64][72];
  const int bh = blockIdx.y;
  const int n0 = blockIdx.x * 64;
  const int r = threadIdx.x >> 2;          // 0..63
  const int c16 = (threadIdx.x & 3) * 16;  // 0,16,32,48
  const size_t base = (size_t)bh * kSeq * kHS;
  const unsigned short* src = v + base + (size_t)(n0 + r) * kHS + c16;
  uint4 a = ((const uint4*)src)[0];
  uint4 b = ((const uint4*)src)[1];
  *(uint4*)&t[r][c16]     = a;
  *(uint4*)&t[r][c16 + 8] = b;
  __syncthreads();
  union { uint4 q[2]; unsigned short u[16]; } o;
#pragma unroll
  for (int i = 0; i < 16; ++i) o.u[i] = t[c16 + i][r];   // vt[d=r][n=n0+c16+i]
  unsigned short* dst = vt + base + (size_t)r * kSeq + n0 + c16;
  ((uint4*)dst)[0] = o.q[0];
  ((uint4*)dst)[1] = o.q[1];
}

// ---------------------------------------------------------------------------
// Kernel 3: flash attention. Block = 128 threads (4 waves), 64 query rows per
// block (16 per wave), key blocks of 64. K and V^T tiles double-buffered in
// LDS via async copies; rows padded to 72 shorts (144 B) -> 16 B aligned and
// bank-conflict-free for the per-lane ds_load_b128 reads.
// ---------------------------------------------------------------------------
__global__ __launch_bounds__(128) void flash_attn_kernel(
    const unsigned short* __restrict__ qg, const unsigned short* __restrict__ kg,
    const unsigned short* __restrict__ vtg, unsigned short* __restrict__ og) {
  __shared__ __align__(16) unsigned short Kt[2][64][72];   // [buf][j][d]
  __shared__ __align__(16) unsigned short Vt[2][64][72];   // [buf][d][j]
  __shared__ __align__(16) unsigned short Pt[4][16][72];   // per-wave P tile
  const int tid = threadIdx.x;
  const int lane = tid & 31, wave = tid >> 5;
  const int hh = lane >> 4, cc = lane & 15;
  const int bh = blockIdx.y;
  const int q0 = blockIdx.x * 64;
  const size_t base = (size_t)bh * kSeq * kHS;   // same for [H,N,D] and [H,D,N]

  const int jr = tid >> 1, jh = tid & 1;         // cooperative tile staging

  // issue async copies for key-block t into LDS buffer t&1 (8 x b128 / thread)
  auto issue_tile = [&](int t) {
    const int buf = t & 1;
    const size_t jb = (size_t)t * 64;
    const unsigned short* ksrc = kg  + base + (jb + jr) * kHS + jh * 32;
    const unsigned short* vsrc = vtg + base + (size_t)jr * kSeq + jb + jh * 32;
    const unsigned int koff = (unsigned int)(uintptr_t)&Kt[buf][jr][jh * 32];
    const unsigned int voff = (unsigned int)(uintptr_t)&Vt[buf][jr][jh * 32];
#pragma unroll
    for (int i = 0; i < 4; ++i) {
      async_b128(koff + 16u * i, (unsigned long long)(uintptr_t)ksrc + 16ull * i);
      async_b128(voff + 16u * i, (unsigned long long)(uintptr_t)vsrc + 16ull * i);
    }
  };

  // Q rows for this wave, in WMMA A layout, kept in registers (2 K-steps of 32)
  Vec16 Aq[2];
  {
    const unsigned short* qrow = qg + base + (size_t)(q0 + wave * 16 + cc) * kHS;
#pragma unroll
    for (int s = 0; s < 2; ++s) {
      Aq[s].q[0] = *(const uint4*)(qrow + s * 32 + 8 * hh);
      Aq[s].q[1] = *(const uint4*)(qrow + s * 32 + 16 + 8 * hh);
    }
  }

  float mrow[8], lrow[8];
  v8f O[4] = {};
#pragma unroll
  for (int r = 0; r < 8; ++r) { mrow[r] = -3.0e38f; lrow[r] = 0.f; }

  constexpr int kNT = kSeq / 64;   // 64 key blocks
  issue_tile(0);

  for (int t = 0; t < kNT; ++t) {
    const int buf = t & 1;
    __syncthreads();               // nobody still reads the buffer we overwrite
    if (t + 1 < kNT) {
      issue_tile(t + 1);           // stream next tile while computing this one
      asm volatile("s_wait_asynccnt 0x8" ::: "memory");  // tile t landed
    } else {
      asm volatile("s_wait_asynccnt 0x0" ::: "memory");
    }
    __syncthreads();               // tile t visible to all waves

    // S = Q * K^T  (B column n -> K row jb+c*16+cc, contiguous in d)
    v8f S[4];
#pragma unroll
    for (int c = 0; c < 4; ++c) {
      v8f a = {};
#pragma unroll
      for (int s = 0; s < 2; ++s) {
        Vec16 bk;
        bk.q[0] = *(const uint4*)&Kt[buf][c * 16 + cc][s * 32 + 8 * hh];
        bk.q[1] = *(const uint4*)&Kt[buf][c * 16 + cc][s * 32 + 16 + 8 * hh];
        a = wmma_bf16(Aq[s].v, bk.v, a);
      }
      S[c] = a;
    }

    // online softmax: row stats per (half, r); columns reduced across 16 lanes
    float scale[8];
#pragma unroll
    for (int r = 0; r < 8; ++r) {
      float m = fmaxf(fmaxf(S[0][r], S[1][r]), fmaxf(S[2][r], S[3][r]));
      m = fmaxf(m, __shfl_xor(m, 1, 32));
      m = fmaxf(m, __shfl_xor(m, 2, 32));
      m = fmaxf(m, __shfl_xor(m, 4, 32));
      m = fmaxf(m, __shfl_xor(m, 8, 32));
      const float mn = fmaxf(mrow[r], m);
      scale[r] = exp2f((mrow[r] - mn) * kLog2e);
      mrow[r] = mn;
    }
#pragma unroll
    for (int r = 0; r < 8; ++r) {
      float rs = 0.f;
#pragma unroll
      for (int c = 0; c < 4; ++c) {
        const float p = exp2f((S[c][r] - mrow[r]) * kLog2e);
        S[c][r] = p;
        rs += p;
      }
      rs += __shfl_xor(rs, 1, 32);
      rs += __shfl_xor(rs, 2, 32);
      rs += __shfl_xor(rs, 4, 32);
      rs += __shfl_xor(rs, 8, 32);
      lrow[r] = lrow[r] * scale[r] + rs;
#pragma unroll
      for (int c = 0; c < 4; ++c) O[c][r] *= scale[r];
      // accumulator layout -> A layout via LDS round trip
#pragma unroll
      for (int c = 0; c < 4; ++c)
        Pt[wave][r + 8 * hh][c * 16 + cc] = f2bf(S[c][r]);
    }
    __syncthreads();

    // O += P * V  (B from V^T tile: row d = dc*16+cc, contiguous in j)
#pragma unroll
    for (int s = 0; s < 2; ++s) {
      Vec16 ap;
      ap.q[0] = *(const uint4*)&Pt[wave][cc][s * 32 + 8 * hh];
      ap.q[1] = *(const uint4*)&Pt[wave][cc][s * 32 + 16 + 8 * hh];
#pragma unroll
      for (int dc = 0; dc < 4; ++dc) {
        Vec16 bv;
        bv.q[0] = *(const uint4*)&Vt[buf][dc * 16 + cc][s * 32 + 8 * hh];
        bv.q[1] = *(const uint4*)&Vt[buf][dc * 16 + cc][s * 32 + 16 + 8 * hh];
        O[dc] = wmma_bf16(ap.v, bv.v, O[dc]);
      }
    }
  }

  // normalize and write attention output as bf16 [B, N, C] for the proj GEMM
  const int b = bh / kHeads, h = bh % kHeads;
#pragma unroll
  for (int r = 0; r < 8; ++r) {
    const float rinv = 1.0f / lrow[r];
    const int i = q0 + wave * 16 + r + 8 * hh;
    unsigned short* dst = og + ((size_t)(b * kSeq + i)) * kDim + h * kHS;
#pragma unroll
    for (int dc = 0; dc < 4; ++dc) dst[dc * 16 + cc] = f2bf(O[dc][r] * rinv);
  }
}

// ---------------------------------------------------------------------------
// Kernel 4: output projection, f32 result + bias.
// ---------------------------------------------------------------------------
__global__ __launch_bounds__(256) void proj_gemm_kernel(
    const unsigned short* __restrict__ ab, const unsigned short* __restrict__ wpb,
    const float* __restrict__ bias, float* __restrict__ out) {
  const int lane = threadIdx.x & 31;
  const int wave = threadIdx.x >> 5;
  const int hh = lane >> 4, cc = lane & 15;
  const int m0 = blockIdx.x * 128 + (wave & 3) * 32;
  const int f0 = blockIdx.y * 64 + (wave >> 2) * 32;
  v8f acc[2][2] = {};
  gemm_tile_384(ab, wpb, m0, f0, hh, cc, acc);
#pragma unroll
  for (int ni = 0; ni < 2; ++ni) {
    const int f = f0 + ni * 16 + cc;
    const float bv = bias[f];
#pragma unroll
    for (int mi = 0; mi < 2; ++mi) {
#pragma unroll
      for (int r = 0; r < 8; ++r) {
        const int m = m0 + mi * 16 + r + 8 * hh;
        out[(size_t)m * kDim + f] = acc[mi][ni][r] + bv;
      }
    }
  }
}

extern "C" void kernel_launch(void* const* d_in, const int* in_sizes, int n_in,
                              void* d_out, int out_size, void* d_ws, size_t ws_size,
                              hipStream_t stream) {
  (void)in_sizes; (void)n_in; (void)out_size; (void)ws_size;
  const float* x      = (const float*)d_in[0];
  const float* w_qkv  = (const float*)d_in[1];
  const float* b_qkv  = (const float*)d_in[2];
  const float* w_proj = (const float*)d_in[3];
  const float* b_proj = (const float*)d_in[4];
  float* out = (float*)d_out;

  // workspace layout (all sizes multiples of 16 B): ~37.5 MB total
  char* ws = (char*)d_ws;
  const size_t nX  = (size_t)kM * kDim;                     // 3,145,728
  const size_t nWq = (size_t)kF * kDim;                     //   442,368
  const size_t nWp = (size_t)kDim * kDim;                   //   147,456
  const size_t nQ  = (size_t)kBatch * kHeads * kSeq * kHS;  // 3,145,728
  unsigned short* xb  = (unsigned short*)ws; ws += nX  * 2;
  unsigned short* wqb = (unsigned short*)ws; ws += nWq * 2;
  unsigned short* wpb = (unsigned short*)ws; ws += nWp * 2;
  unsigned short* qb  = (unsigned short*)ws; ws += nQ  * 2;
  unsigned short* kb  = (unsigned short*)ws; ws += nQ  * 2;
  unsigned short* vb  = (unsigned short*)ws; ws += nQ  * 2;
  unsigned short* vtb = (unsigned short*)ws; ws += nQ  * 2;
  unsigned short* atb = (unsigned short*)ws; ws += nX  * 2;

  cvt_bf16_kernel<<<(int)((nX  + 255) / 256), 256, 0, stream>>>(x,      xb,  (int)nX);
  cvt_bf16_kernel<<<(int)((nWq + 255) / 256), 256, 0, stream>>>(w_qkv,  wqb, (int)nWq);
  cvt_bf16_kernel<<<(int)((nWp + 255) / 256), 256, 0, stream>>>(w_proj, wpb, (int)nWp);

  dim3 g1(kM / 128, kF / 64);          // 64 x 18
  qkv_gemm_kernel<<<g1, 256, 0, stream>>>(xb, wqb, b_qkv, qb, kb, vb);

  dim3 gt(kSeq / 64, kBatch * kHeads); // 64 x 12
  vtrans_kernel<<<gt, 256, 0, stream>>>(vb, vtb);

  dim3 g2(kSeq / 64, kBatch * kHeads); // 64 x 12
  flash_attn_kernel<<<g2, 128, 0, stream>>>(qb, kb, vtb, atb);

  dim3 g3(kM / 128, kDim / 64);        // 64 x 6
  proj_gemm_kernel<<<g3, 256, 0, stream>>>(atb, wpb, b_proj, out);
}
